// GlobalLayer_36910948942525
// MI455X (gfx1250) — matrix operand order repaired and checked by reading, not compiled, not run on hardware
//
#include <hip/hip_runtime.h>
#include <hip/hip_bf16.h>

typedef __attribute__((ext_vector_type(16))) _Float16 v16h;
typedef __attribute__((ext_vector_type(8)))  _Float16 v8h;
typedef __attribute__((ext_vector_type(8)))  float    v8f;

#define N_TOK 2048
#define RANK1 1707u   // kth = N-1-N//6 = 1706 -> smallest key with count(<=key) >= 1707

// ---------- helpers ----------
__device__ __forceinline__ unsigned fkey(float f) {           // order-preserving f32->u32
  unsigned u = __float_as_uint(f);
  return (u & 0x80000000u) ? ~u : (u | 0x80000000u);
}
__device__ __forceinline__ float kinv(unsigned k) {           // inverse of fkey
  unsigned u = (k & 0x80000000u) ? (k & 0x7FFFFFFFu) : ~k;
  return __uint_as_float(u);
}
__device__ __forceinline__ int wredi(int x) {
  #pragma unroll
  for (int o = 16; o; o >>= 1) x += __shfl_xor(x, o, 32);
  return x;
}
__device__ __forceinline__ float wredf(float x) {
  #pragma unroll
  for (int o = 16; o; o >>= 1) x += __shfl_xor(x, o, 32);
  return x;
}
__device__ __forceinline__ float wmaxf(float x) {
  #pragma unroll
  for (int o = 16; o; o >>= 1) x = fmaxf(x, __shfl_xor(x, o, 32));
  return x;
}
__device__ __forceinline__ v16h pack16(v8h lo, v8h hi) {
  v16h r;
  #pragma unroll
  for (int i = 0; i < 8; i++) { r[i] = lo[i]; r[8 + i] = hi[i]; }
  return r;
}
// A fragment (16x32 f16): lane<16 -> M=lane, k in {0..7,16..23}; lane>=16 -> k in {8..15,24..31}
__device__ __forceinline__ v16h ldA(const _Float16* rowBase, int half) {
  const _Float16* p = rowBase + 8 * half;
  return pack16(*(const v8h*)p, *(const v8h*)(p + 16));
}
// B fragment (32x16 f16): lane<16 -> N=lane, K in {0..7,16..23}; lane>=16 -> K in {8..15,24..31}
__device__ __forceinline__ v16h ldB(const _Float16* colBase, int half) {
  const _Float16* p = colBase + 8 * half;
  return pack16(*(const v8h*)p, *(const v8h*)(p + 16));
}
__device__ __forceinline__ v8f wmma_f16(v16h a, v16h b, v8f c) {
  return __builtin_amdgcn_wmma_f32_16x16x32_f16(false, a, false, b, (short)0, c, false, false);
}

// ---------- prep kernels ----------
// coor_embed = leaky_relu(res_coor @ embed_W.T + embed_b, 0.1)   (N,30)
__global__ __launch_bounds__(256) void prep_cembed(const float* __restrict__ rc,
                                                   const float* __restrict__ eW,
                                                   const float* __restrict__ eb,
                                                   float* __restrict__ ce) {
  int n = blockIdx.x * 256 + threadIdx.x;
  if (n >= N_TOK) return;
  float x = rc[n * 3 + 0], y = rc[n * 3 + 1], z = rc[n * 3 + 2];
  #pragma unroll
  for (int j = 0; j < 30; j++) {
    float s = eW[j * 3 + 0] * x + eW[j * 3 + 1] * y + eW[j * 3 + 2] * z + eb[j];
    ce[n * 30 + j] = (s > 0.f) ? s : 0.1f * s;
  }
}

// Per (b,n): h_embed = h + ce ; qk = h_embed @ qk_W.T + qk_b (60) split even->Q, odd->K.
// Store Q (scale 5^-0.5 folded) and K as f16 padded 30->32; also H^T (b,32,N) f16 for WMMA B.
__global__ __launch_bounds__(256) void prep_qkh(const float* __restrict__ h,
                                                const float* __restrict__ ce,
                                                const float* __restrict__ qkW,
                                                const float* __restrict__ qkb,
                                                _Float16* __restrict__ Q,
                                                _Float16* __restrict__ Km,
                                                _Float16* __restrict__ HT) {
  __shared__ float sW[60 * 30];
  __shared__ float sB[60];
  for (int i = threadIdx.x; i < 1800; i += 256) sW[i] = qkW[i];
  for (int i = threadIdx.x; i < 60;   i += 256) sB[i] = qkb[i];
  __syncthreads();
  int idx = blockIdx.x * 256 + threadIdx.x;          // b*2048 + n
  if (idx >= 32 * N_TOK) return;
  int b = idx >> 11, n = idx & (N_TOK - 1);
  float he[30], hr[30];
  #pragma unroll
  for (int d = 0; d < 30; d++) {
    float hv = h[(size_t)idx * 30 + d];
    hr[d] = hv;
    he[d] = hv + ce[n * 30 + d];
  }
  _Float16* qrow = Q  + (size_t)idx * 32;
  _Float16* krow = Km + (size_t)idx * 32;
  const float qscale = 0.44721359549995793f;          // 1/sqrt(HEAD_D=5)
  #pragma unroll 2
  for (int j = 0; j < 30; j++) {
    float aq = sB[2 * j], ak = sB[2 * j + 1];
    const float* wq = sW + (2 * j) * 30;
    const float* wk = sW + (2 * j + 1) * 30;
    #pragma unroll
    for (int d = 0; d < 30; d++) { aq += wq[d] * he[d]; ak += wk[d] * he[d]; }
    qrow[j] = (_Float16)(aq * qscale);
    krow[j] = (_Float16)ak;
  }
  qrow[30] = (_Float16)0.f; qrow[31] = (_Float16)0.f;
  krow[30] = (_Float16)0.f; krow[31] = (_Float16)0.f;
  #pragma unroll
  for (int d = 0; d < 30; d++)
    HT[((size_t)b * 32 + d) * N_TOK + n] = (_Float16)hr[d];
  HT[((size_t)b * 32 + 30) * N_TOK + n] = (_Float16)0.f;
  HT[((size_t)b * 32 + 31) * N_TOK + n] = (_Float16)0.f;
}

// WpT[n][k] = weight[k][n] for k<30 else 0   (64x32 f16)
__global__ __launch_bounds__(256) void prep_w(const float* __restrict__ weight,
                                              _Float16* __restrict__ WpT) {
  int i = blockIdx.x * 256 + threadIdx.x;             // n*32 + k
  if (i >= 64 * 32) return;
  int n = i >> 5, k = i & 31;
  WpT[i] = (k < 30) ? (_Float16)weight[k * 64 + n] : (_Float16)0.f;
}

// ---------- fused attention kernel ----------
// One workgroup (128 thr = 4 waves) owns 16 query rows of one batch.
// LDS: dots stripe 16x2048 f32 (128KB) + partials 4x16x32 + outF 16x32
extern "C" __global__ __launch_bounds__(128) void attn_fused(
    const _Float16* __restrict__ Q, const _Float16* __restrict__ Km,
    const _Float16* __restrict__ HT, const _Float16* __restrict__ WpT,
    const float* __restrict__ bias, float* __restrict__ out) {
  extern __shared__ char smem[];
  float* dotsS = (float*)smem;                                   // [16][2048]
  float* partS = (float*)(smem + 16 * N_TOK * 4);                // [4][16][32]
  float* outFS = (float*)(smem + 16 * N_TOK * 4 + 4 * 16 * 32 * 4); // [16][32]

  const int tid  = threadIdx.x;
  const int lane = tid & 31, wave = tid >> 5;
  const int half = lane >> 4, l16 = lane & 15;
  const int batch = blockIdx.x >> 7;
  const int row0  = (blockIdx.x & 127) * 16;

  // ---- Phase 1: dots = (Q*scale) @ K^T, 128 column tiles split over 4 waves
  v16h aq = ldA(Q + ((size_t)batch * N_TOK + row0 + l16) * 32, half);
  for (int jt = wave; jt < 128; jt += 4) {
    v16h bk = ldB(Km + ((size_t)batch * N_TOK + jt * 16 + l16) * 32, half);
    v8f c = {};
    c = wmma_f16(aq, bk, c);
    int col = jt * 16 + l16;
    #pragma unroll
    for (int r = 0; r < 8; r++) dotsS[(r + 8 * half) * N_TOK + col] = c[r];
  }
  __syncthreads();

  // ---- Phase 2: exact order-statistic threshold + softmax(x/0.3), 1 wave : 4 rows
  for (int rr = 0; rr < 4; rr++) {
    int row = wave * 4 + rr;
    float* rowp = dotsS + row * N_TOK;
    unsigned kk[64];
    #pragma unroll
    for (int t = 0; t < 64; t++) kk[t] = fkey(rowp[t * 32 + lane]);
    // binary search: smallest key with count(<=key) >= RANK1
    unsigned klo = 0u, khi = 0xFFFFFFFFu;
    while (klo < khi) {
      unsigned mid = klo + ((khi - klo) >> 1);
      int cnt = 0;
      #pragma unroll
      for (int t = 0; t < 64; t++) cnt += (kk[t] <= mid) ? 1 : 0;
      cnt = wredi(cnt);
      if ((unsigned)cnt >= RANK1) khi = mid; else klo = mid + 1u;
    }
    const unsigned tk = klo;
    // masked max
    float m = -3.4e38f;
    #pragma unroll
    for (int t = 0; t < 64; t++) {
      float v  = kinv(kk[t]);
      float vm = (kk[t] < tk) ? -1e-7f : v;   // dots < threshold  <=>  key < tk
      m = fmaxf(m, vm);
    }
    m = wmaxf(m);
    // exp + sum
    float s = 0.f;
    #pragma unroll
    for (int t = 0; t < 64; t++) {
      float v  = kinv(kk[t]);
      float vm = (kk[t] < tk) ? -1e-7f : v;
      float e  = __expf((vm - m) * (1.0f / 0.3f));
      s += e;
      rowp[t * 32 + lane] = e;
    }
    s = wredf(s);
    float inv = 1.0f / s;
    #pragma unroll
    for (int t = 0; t < 64; t++) rowp[t * 32 + lane] *= inv;
  }
  __syncthreads();

  // ---- Phase 3: outF = attn(16x2048) @ H^T-cols (2048x32), K-chunks split over waves
  v8f acc0 = {}, acc1 = {};
  for (int kt = wave; kt < 64; kt += 4) {
    const float* ap = dotsS + l16 * N_TOK + kt * 32 + 8 * half;
    v16h a;
    #pragma unroll
    for (int i = 0; i < 8; i++) { a[i] = (_Float16)ap[i]; a[8 + i] = (_Float16)ap[16 + i]; }
    v16h b0 = ldB(HT + ((size_t)batch * 32 + l16)      * N_TOK + kt * 32, half);
    v16h b1 = ldB(HT + ((size_t)batch * 32 + 16 + l16) * N_TOK + kt * 32, half);
    acc0 = wmma_f16(a, b0, acc0);
    acc1 = wmma_f16(a, b1, acc1);
  }
  #pragma unroll
  for (int r = 0; r < 8; r++) {
    partS[(wave * 16 + r + 8 * half) * 32 + l16]      = acc0[r];
    partS[(wave * 16 + r + 8 * half) * 32 + 16 + l16] = acc1[r];
  }
  __syncthreads();
  for (int e = tid; e < 512; e += 128) {
    int mm = e >> 5, dd = e & 31;
    outFS[e] = partS[(0 * 16 + mm) * 32 + dd] + partS[(1 * 16 + mm) * 32 + dd] +
               partS[(2 * 16 + mm) * 32 + dd] + partS[(3 * 16 + mm) * 32 + dd];
  }
  __syncthreads();

  // ---- Phase 4: out = outF(16x32) @ WpT^T(32x64) + bias, 1 wave : 16 output cols
  {
    const float* ap = outFS + l16 * 32 + 8 * half;
    v16h a;
    #pragma unroll
    for (int i = 0; i < 8; i++) { a[i] = (_Float16)ap[i]; a[8 + i] = (_Float16)ap[16 + i]; }
    int ncol = wave * 16 + l16;
    v16h b = ldB(WpT + ncol * 32, half);
    v8f c = {};
    c = wmma_f16(a, b, c);
    float bv = bias[ncol];
    #pragma unroll
    for (int r = 0; r < 8; r++) {
      int m = r + 8 * half;
      out[((size_t)batch * N_TOK + row0 + m) * 64 + ncol] = c[r] + bv;
    }
  }
}

// ---------- launch ----------
extern "C" void kernel_launch(void* const* d_in, const int* in_sizes, int n_in,
                              void* d_out, int out_size, void* d_ws, size_t ws_size,
                              hipStream_t stream) {
  (void)in_sizes; (void)n_in; (void)out_size; (void)ws_size;
  const float* h      = (const float*)d_in[0];
  const float* rc     = (const float*)d_in[1];
  const float* eW     = (const float*)d_in[2];
  const float* eb     = (const float*)d_in[3];
  const float* qkW    = (const float*)d_in[4];
  const float* qkb    = (const float*)d_in[5];
  const float* weight = (const float*)d_in[6];
  const float* bias   = (const float*)d_in[7];
  float* out = (float*)d_out;

  char* ws = (char*)d_ws;
  float*    ce  = (float*)ws;                                   // 2048*30*4 = 245760 B
  _Float16* Q   = (_Float16*)(ws + 245760);                     // 32*2048*32*2 = 4 MiB
  _Float16* Km  = (_Float16*)(ws + 245760 + 4194304);           // 4 MiB
  _Float16* HT  = (_Float16*)(ws + 245760 + 2 * 4194304);       // 4 MiB
  _Float16* WpT = (_Float16*)(ws + 245760 + 3 * 4194304);       // 4 KiB

  prep_cembed<<<8, 256, 0, stream>>>(rc, eW, eb, ce);
  prep_w<<<8, 256, 0, stream>>>(weight, WpT);
  prep_qkh<<<256, 256, 0, stream>>>(h, ce, qkW, qkb, Q, Km, HT);

  size_t shmem = (size_t)16 * N_TOK * 4 + 4 * 16 * 32 * 4 + 16 * 32 * 4; // 141312 B
  attn_fused<<<32 * 128, 128, shmem, stream>>>(Q, Km, HT, WpT, bias, out);
}
